// GCN_Classification_87771951661435
// MI455X (gfx1250) — matrix operand order repaired and checked by reading, not compiled, not run on hardware
//
#include <hip/hip_runtime.h>

#define N_NODES 50000
#define N_EDGES 800000
#define FEAT 256
#define N_GRAPHS 512
#define MTILE 80                 // 50000 = 625 * 80; 5 row-tiles of 16 per block

typedef __attribute__((ext_vector_type(16))) __bf16 v16bf;
typedef __attribute__((ext_vector_type(8)))  float  v8f;

__device__ __forceinline__ unsigned short f2bf(float f) {
    unsigned u = __float_as_uint(f);
    unsigned rb = ((u >> 16) & 1u) + 0x7FFFu;   // round-to-nearest-even
    return (unsigned short)((u + rb) >> 16);
}

__device__ __forceinline__ void atomAddF32(float* p, float v) {
    __hip_atomic_fetch_add(p, v, __ATOMIC_RELAXED, __HIP_MEMORY_SCOPE_AGENT);
}

// ---------------- degree / normalization ----------------
__global__ __launch_bounds__(256) void deg_count_k(const long long* __restrict__ ei,
                                                   float* __restrict__ deg) {
    int e = blockIdx.x * 256 + threadIdx.x;       // grid sized exactly to N_EDGES
    atomAddF32(&deg[(int)ei[N_EDGES + e]], 1.0f);
}

__global__ __launch_bounds__(256) void dinv_k(float* __restrict__ deg_dinv) {
    int i = blockIdx.x * 256 + threadIdx.x;
    if (i < N_NODES) deg_dinv[i] = rsqrtf(deg_dinv[i] + 1.0f);
}

// ---------------- weight convert: W[k][n] f32 -> Wt[n][k] bf16 ----------------
__global__ __launch_bounds__(256) void conv_wt_k(const float* __restrict__ W,
                                                 unsigned short* __restrict__ Wt) {
    int idx = blockIdx.x * 256 + threadIdx.x;     // 65536 = 256 blocks
    int k = idx >> 8, n = idx & 255;
    Wt[n * 256 + k] = f2bf(W[idx]);
}

// ---------------- WMMA GEMM: C[M,256] = A[M,256] * W[256,256] ----------------
// Block: 80 rows (5 row-tiles) x all 256 cols. 8 waves; wave w owns cols [32w,32w+32).
// B slab (256x32 bf16) filled by GLOBAL_LOAD_ASYNC_TO_LDS_B128 (ASYNCcnt);
// A slab converted f32->bf16 through VALU (overlaps with async B fill).
__global__ __launch_bounds__(256) void gemm_wmma_k(const float* __restrict__ A,
                                                   const unsigned short* __restrict__ Wt,
                                                   float* __restrict__ C) {
    __shared__ unsigned short Asl[MTILE][34];     // 17-dword row stride: conflict-free
    __shared__ unsigned short Bsl[256][40];       // 80B row stride: 16B-aligned, conflict-free

    const int tid  = threadIdx.x;
    const int wave = tid >> 5;
    const int lane = tid & 31;
    const int hi   = lane >> 4;
    const int l15  = lane & 15;
    const int mbase = blockIdx.x * MTILE;
    const int n0    = wave * 32;
    const unsigned long long wbase = (unsigned long long)Wt;

    v8f acc[10];
    #pragma unroll
    for (int i = 0; i < 10; ++i) acc[i] = {};

    for (int kb = 0; kb < 256; kb += 32) {
        // async B slab fill: 256 rows x 64B = 1024 x 16B chunks, 4 per thread
        #pragma unroll
        for (int j = 0; j < 4; ++j) {
            int c = j * 256 + tid;
            int n = c >> 2, part = c & 3;
            unsigned ldsa = (unsigned)(size_t)&Bsl[n][part * 8];
            unsigned voff = (unsigned)(n * 512 + kb * 2 + part * 16);
            asm volatile("global_load_async_to_lds_b128 %0, %1, %2 offset:0"
                         :: "v"(ldsa), "v"(voff), "s"(wbase) : "memory");
        }
        // A slab: 80 rows x 32 k f32 -> bf16 (10 elements / thread)
        #pragma unroll
        for (int r = 0; r < 10; ++r) {
            int idx = tid + r * 256;
            int row = idx >> 5, k = idx & 31;
            Asl[row][k] = f2bf(A[(size_t)(mbase + row) * 256 + kb + k]);
        }
        asm volatile("s_wait_asynccnt 0x0" ::: "memory");
        __syncthreads();

        // B fragments once per k-step (contiguous k, same as A: lanes0-15 K0..15, lanes16-31 K16..31)
        union { unsigned u[8]; v16bf v; } b0, b1;
        #pragma unroll
        for (int i = 0; i < 8; ++i) {
            int kbf = 2 * i + hi * 16;
            b0.u[i] = *(const unsigned*)&Bsl[n0 + l15][kbf];
            b1.u[i] = *(const unsigned*)&Bsl[n0 + 16 + l15][kbf];
        }
        // A fragments per row-tile (lanes0-15 K0..15, lanes16-31 K16..31)
        #pragma unroll
        for (int mt = 0; mt < 5; ++mt) {
            union { unsigned u[8]; v16bf v; } a;
            #pragma unroll
            for (int i = 0; i < 8; ++i) {
                int ka = 2 * i + hi * 16;
                a.u[i] = *(const unsigned*)&Asl[mt * 16 + l15][ka];
            }
            acc[mt * 2 + 0] = __builtin_amdgcn_wmma_f32_16x16x32_bf16(
                false, a.v, false, b0.v, (short)0, acc[mt * 2 + 0], false, false);
            acc[mt * 2 + 1] = __builtin_amdgcn_wmma_f32_16x16x32_bf16(
                false, a.v, false, b1.v, (short)0, acc[mt * 2 + 1], false, false);
        }
        __syncthreads();
    }

    // C f32 layout: VGPR r -> row r + 8*hi, col = lane&15
    #pragma unroll
    for (int mt = 0; mt < 5; ++mt) {
        #pragma unroll
        for (int r = 0; r < 8; ++r) {
            size_t row = (size_t)(mbase + mt * 16 + r + hi * 8);
            C[row * 256 + n0 + l15]      = acc[mt * 2 + 0][r];
            C[row * 256 + n0 + 16 + l15] = acc[mt * 2 + 1][r];
        }
    }
}

// ---------------- agg = h * dinv^2 + bias ----------------
__global__ __launch_bounds__(256) void agg_init_k(const float* __restrict__ h,
                                                  const float* __restrict__ dinv,
                                                  const float* __restrict__ bias,
                                                  float* __restrict__ agg) {
    size_t idx = (size_t)blockIdx.x * 256 + threadIdx.x;   // 12.8M exact
    int i = (int)(idx >> 8), f = (int)(idx & 255);
    float d = dinv[i];
    agg[idx] = h[idx] * d * d + bias[f];
}

// ---------------- edge scatter: agg[dst] += h[src] * dinv[src]*dinv[dst] ----------------
__global__ __launch_bounds__(64) void edge_scatter_k(const long long* __restrict__ ei,
                                                     const float* __restrict__ dinv,
                                                     const float* __restrict__ h,
                                                     float* __restrict__ agg) {
    int e = blockIdx.x;
    int s = (int)ei[e];
    int d = (int)ei[N_EDGES + e];
    float ne = dinv[s] * dinv[d];
    int f = threadIdx.x * 4;
    float4 hv = *(const float4*)(h + (size_t)s * 256 + f);
    float* base = agg + (size_t)d * 256 + f;
    atomAddF32(base + 0, hv.x * ne);
    atomAddF32(base + 1, hv.y * ne);
    atomAddF32(base + 2, hv.z * ne);
    atomAddF32(base + 3, hv.w * ne);
}

__global__ __launch_bounds__(256) void relu_k(float* __restrict__ a) {
    size_t idx = (size_t)blockIdx.x * 256 + threadIdx.x;
    a[idx] = fmaxf(a[idx], 0.0f);
}

// ---------------- pooling ----------------
__global__ __launch_bounds__(64) void pool_sum_k(const float* __restrict__ h,
                                                 const long long* __restrict__ batch,
                                                 float* __restrict__ pooled,
                                                 float* __restrict__ counts) {
    int i = blockIdx.x;
    int g = (int)batch[i];
    if (threadIdx.x == 0) atomAddF32(&counts[g], 1.0f);
    int f = threadIdx.x * 4;
    float4 hv = *(const float4*)(h + (size_t)i * 256 + f);
    float* base = pooled + (size_t)g * 256 + f;
    atomAddF32(base + 0, hv.x);
    atomAddF32(base + 1, hv.y);
    atomAddF32(base + 2, hv.z);
    atomAddF32(base + 3, hv.w);
}

__global__ __launch_bounds__(256) void final_linear_k(const float* __restrict__ pooled,
                                                      const float* __restrict__ counts,
                                                      const float* __restrict__ Wl,
                                                      const float* __restrict__ bl,
                                                      float* __restrict__ out) {
    int idx = blockIdx.x * 256 + threadIdx.x;
    if (idx >= N_GRAPHS * 2) return;
    int g = idx >> 1, c = idx & 1;
    float inv = 1.0f / fmaxf(counts[g], 1.0f);
    float acc = bl[c];
    const float* p = pooled + (size_t)g * 256;
    #pragma unroll 4
    for (int f = 0; f < 256; ++f) acc += p[f] * inv * Wl[f * 2 + c];
    out[idx] = acc;
}

extern "C" void kernel_launch(void* const* d_in, const int* in_sizes, int n_in,
                              void* d_out, int out_size, void* d_ws, size_t ws_size,
                              hipStream_t stream) {
    const float*     x     = (const float*)d_in[0];
    const long long* ei    = (const long long*)d_in[1];
    const long long* batch = (const long long*)d_in[2];
    const float* Ws[3] = { (const float*)d_in[3], (const float*)d_in[5], (const float*)d_in[7] };
    const float* bs[3] = { (const float*)d_in[4], (const float*)d_in[6], (const float*)d_in[8] };
    const float* Wl = (const float*)d_in[9];
    const float* bl = (const float*)d_in[10];
    float* out = (float*)d_out;

    auto al = [](size_t v) { return (v + 255) & ~(size_t)255; };
    char* w = (char*)d_ws;
    size_t o = 0;
    float*          dinv   = (float*)(w + o);          o += al((size_t)N_NODES * 4);
    unsigned short* Wt     = (unsigned short*)(w + o); o += al((size_t)256 * 256 * 2);
    float*          hbuf   = (float*)(w + o);          o += al((size_t)N_NODES * FEAT * 4);
    float*          aggbuf = (float*)(w + o);          o += al((size_t)N_NODES * FEAT * 4);
    float*          pooled = (float*)(w + o);          o += al((size_t)N_GRAPHS * FEAT * 4);
    float*          counts = (float*)(w + o);          o += al((size_t)N_GRAPHS * 4);
    (void)o; (void)ws_size; (void)in_sizes; (void)n_in; (void)out_size;

    // degree + symmetric normalization
    hipMemsetAsync(dinv, 0, (size_t)N_NODES * 4, stream);
    deg_count_k<<<N_EDGES / 256, 256, 0, stream>>>(ei, dinv);
    dinv_k<<<(N_NODES + 255) / 256, 256, 0, stream>>>(dinv);

    const float* in = x;
    for (int l = 0; l < 3; ++l) {
        conv_wt_k<<<256, 256, 0, stream>>>(Ws[l], Wt);
        gemm_wmma_k<<<N_NODES / MTILE, 256, 0, stream>>>(in, Wt, hbuf);
        agg_init_k<<<N_NODES, 256, 0, stream>>>(hbuf, dinv, bs[l], aggbuf);
        edge_scatter_k<<<N_EDGES, 64, 0, stream>>>(ei, dinv, hbuf, aggbuf);
        relu_k<<<N_NODES, 256, 0, stream>>>(aggbuf);
        in = aggbuf;
    }

    // global mean pool + linear head
    hipMemsetAsync(pooled, 0, (size_t)N_GRAPHS * FEAT * 4, stream);
    hipMemsetAsync(counts, 0, (size_t)N_GRAPHS * 4, stream);
    pool_sum_k<<<N_NODES, 64, 0, stream>>>(aggbuf, batch, pooled, counts);
    final_linear_k<<<(N_GRAPHS * 2 + 255) / 256, 256, 0, stream>>>(pooled, counts, Wl, bl, out);
}
